// TransformerDecoder_28656021799400
// MI455X (gfx1250) — matrix operand order repaired and checked by reading, not compiled, not run on hardware
//
#include <hip/hip_runtime.h>
#include <hip/hip_bf16.h>
#include <math.h>

// Model dims (compile-time constants from the reference)
#define NB   8
#define TT   512
#define VV   32000
#define DD   512
#define DIN  1024
#define HH   8
#define LL   2
#define FF   2048
#define HD   64

typedef __attribute__((ext_vector_type(16))) __bf16 v16bf;
typedef __attribute__((ext_vector_type(8)))  float  v8f;

__device__ __forceinline__ v8f wmma_bf16(v16bf a, v16bf b, v8f c) {
  return __builtin_amdgcn_wmma_f32_16x16x32_bf16(false, a, false, b, (short)0, c, false, false);
}

// ---------------------------------------------------------------------------
// Embedding + positional table gather: x[n*T+t, d] = emb[cap[n,t], d] + pos[t, d]
// ---------------------------------------------------------------------------
__global__ __launch_bounds__(256) void embed_kernel(const int* __restrict__ cap,
                                                    const float* __restrict__ emb,
                                                    const float* __restrict__ pos,
                                                    float* __restrict__ x) {
  int idx = blockIdx.x * blockDim.x + threadIdx.x;
  if (idx >= NB * TT * DD) return;
  int d  = idx % DD;
  int nt = idx / DD;
  int t  = nt % TT;
  x[idx] = emb[(size_t)cap[nt] * DD + d] + pos[(size_t)t * DD + d];
}

// ---------------------------------------------------------------------------
// Small dense GEMM (few rows): C[M,Nc] = A[M,K] @ B[K,Nc] + bias
// ---------------------------------------------------------------------------
__global__ __launch_bounds__(256) void smallmm_kernel(const float* __restrict__ A,
                                                      const float* __restrict__ B,
                                                      const float* __restrict__ bias,
                                                      float* __restrict__ C,
                                                      int M, int K, int Nc) {
  int idx = blockIdx.x * blockDim.x + threadIdx.x;
  if (idx >= M * Nc) return;
  int m = idx / Nc, n = idx % Nc;
  float s = bias ? bias[n] : 0.0f;
  const float* a = A + (size_t)m * K;
  for (int k = 0; k < K; ++k) s += a[k] * B[(size_t)k * Nc + n];
  C[idx] = s;
}

// ---------------------------------------------------------------------------
// Pack fp32 A[M,K] -> bf16 WMMA A-fragment layout.
// Ap[((mt*Kt + kt)*32 + lane)*16 + e]; lane = h*16+m:
//   e in 0..15 : A[mt*16+m][kt*32 + h*16 + e]
// ---------------------------------------------------------------------------
__global__ __launch_bounds__(256) void packA_kernel(const float* __restrict__ A,
                                                    __bf16* __restrict__ Ap,
                                                    int Mt, int Kt, int K) {
  int idx = blockIdx.x * blockDim.x + threadIdx.x;
  if (idx >= Mt * Kt * 32) return;
  int lane = idx & 31, tile = idx >> 5;
  int kt = tile % Kt, mt = tile / Kt;
  int h = lane >> 4, m = lane & 15;
  const float* ar = A + (size_t)(mt * 16 + m) * K + kt * 32;
  __bf16* dst = Ap + (size_t)idx * 16;
  #pragma unroll
  for (int e = 0; e < 8; ++e) dst[e]     = (__bf16)ar[h * 16 + e];
  #pragma unroll
  for (int e = 0; e < 8; ++e) dst[8 + e] = (__bf16)ar[h * 16 + 8 + e];
}

// ---------------------------------------------------------------------------
// Pack fp32 B[K,Nc] -> bf16 WMMA B-fragment layout.
// Bp[((nt*Kt + kt)*32 + lane)*16 + e]; lane = h*16+n:
//   e in 0..15 : B[kt*32 + h*16 + e][nt*16 + n]
// ---------------------------------------------------------------------------
__global__ __launch_bounds__(256) void packB_kernel(const float* __restrict__ B,
                                                    __bf16* __restrict__ Bp,
                                                    int Nt, int Kt, int Nc) {
  int idx = blockIdx.x * blockDim.x + threadIdx.x;
  if (idx >= Nt * Kt * 32) return;
  int lane = idx & 31, tile = idx >> 5;
  int kt = tile % Kt, nt = tile / Kt;
  int h = lane >> 4, n = lane & 15;
  const float* bp = B + (size_t)(kt * 32 + h * 16) * Nc + nt * 16 + n;
  __bf16* dst = Bp + (size_t)idx * 16;
  #pragma unroll
  for (int e = 0; e < 16; ++e) dst[e] = (__bf16)bp[(size_t)e * Nc];
}

// ---------------------------------------------------------------------------
// WMMA GEMM: C = unpack(Ap) @ unpack(Bp) + bias, optional relu.
// One wave -> 32 x 64 tile of C: 2 A frags x 4 B frags = 8 WMMA per K step,
// each B fragment reused twice (192 B loaded per 8 WMMA).
// Prefetch B stream 4 K-steps ahead (global_prefetch_b8) to prime L2.
// ---------------------------------------------------------------------------
__global__ __launch_bounds__(256) void gemm_kernel(const __bf16* __restrict__ Ap,
                                                   const __bf16* __restrict__ Bp,
                                                   const float* __restrict__ bias,
                                                   float* __restrict__ C,
                                                   int Mb, int Nblk, int Kt, int Nc,
                                                   int relu) {
  int gid  = blockIdx.x * blockDim.x + threadIdx.x;
  int wave = gid >> 5, lane = gid & 31;
  if (wave >= Mb * Nblk) return;
  int nb = wave % Nblk, mb = wave / Nblk;

  v8f acc[2][4] = {};
  const __bf16* aBase0 = Ap + (((size_t)(mb * 2)     * Kt * 32) + lane) * 16;
  const __bf16* aBase1 = Ap + (((size_t)(mb * 2 + 1) * Kt * 32) + lane) * 16;
  for (int kt = 0; kt < Kt; ++kt) {
    v16bf a0 = *(const v16bf*)(aBase0 + (size_t)kt * 512);
    v16bf a1 = *(const v16bf*)(aBase1 + (size_t)kt * 512);
    if (kt + 4 < Kt) {
      const __bf16* pf = Bp + ((((size_t)(nb * 4) * Kt + (kt + 4)) * 32) + lane) * 16;
      __builtin_prefetch(pf, 0, 1);
    }
    #pragma unroll
    for (int j = 0; j < 4; ++j) {
      const __bf16* bPtr = Bp + ((((size_t)(nb * 4 + j) * Kt + kt) * 32) + lane) * 16;
      v16bf b = *(const v16bf*)bPtr;
      acc[0][j] = wmma_bf16(a0, b, acc[0][j]);
      acc[1][j] = wmma_bf16(a1, b, acc[1][j]);
    }
  }

  int h = lane >> 4, n2 = lane & 15;
  #pragma unroll
  for (int i = 0; i < 2; ++i) {
    #pragma unroll
    for (int j = 0; j < 4; ++j) {
      int col = (nb * 4 + j) * 16 + n2;
      float bv = bias ? bias[col] : 0.0f;
      #pragma unroll
      for (int v = 0; v < 8; ++v) {
        int row = (mb * 2 + i) * 16 + v + 8 * h;
        float val = acc[i][j][v] + bv;
        if (relu) val = fmaxf(val, 0.0f);
        C[(size_t)row * Nc + col] = val;
      }
    }
  }
}

// ---------------------------------------------------------------------------
// Fused causal self-attention, one wave per (n, head, 16-query tile).
// Phase 1: scores = q k^T / 8 (WMMA, bf16) -> LDS (16 x 512 fp32)
// Phase 2: causal softmax in LDS (warp-parallel)
// Phase 3: out = probs @ v (WMMA, bf16)
// ---------------------------------------------------------------------------
__global__ __launch_bounds__(32) void attn_kernel(const float* __restrict__ q,
                                                  const float* __restrict__ k,
                                                  const float* __restrict__ v,
                                                  float* __restrict__ out) {
  __shared__ float sc[16 * TT];  // 32 KB
  int blk  = blockIdx.x;
  int mt   = blk % (TT / 16);
  int nh   = blk / (TT / 16);
  int head = nh % HH;
  int n    = nh / HH;
  int lane = threadIdx.x;
  int h2 = lane >> 4, l2 = lane & 15;

  // q A-fragments for K=64 (two 32-wide chunks)
  const float* qrow = q + (size_t)(n * TT + mt * 16 + l2) * DD + head * HD;
  v16bf qa[2];
  #pragma unroll
  for (int kt = 0; kt < 2; ++kt) {
    #pragma unroll
    for (int e = 0; e < 8; ++e) qa[kt][e]     = (__bf16)qrow[kt * 32 + h2 * 16 + e];
    #pragma unroll
    for (int e = 0; e < 8; ++e) qa[kt][8 + e] = (__bf16)qrow[kt * 32 + h2 * 16 + 8 + e];
  }

  // Phase 1: scores tiles over all 512 keys
  for (int st = 0; st < TT / 16; ++st) {
    v8f acc = {};
    #pragma unroll
    for (int kt = 0; kt < 2; ++kt) {
      const float* krow = k + (size_t)(n * TT + st * 16 + l2) * DD + head * HD + kt * 32 + h2 * 16;
      v16bf b;
      #pragma unroll
      for (int e = 0; e < 16; ++e) b[e] = (__bf16)krow[e];
      acc = wmma_bf16(qa[kt], b, acc);
    }
    #pragma unroll
    for (int vv = 0; vv < 8; ++vv) {
      int rowl = vv + 8 * h2;
      int tq = mt * 16 + rowl;
      int s  = st * 16 + l2;
      float val = acc[vv] * 0.125f;              // 1/sqrt(64)
      if (s > tq) val = -__builtin_inff();       // causal mask
      sc[rowl * TT + s] = val;
    }
  }
  __syncthreads();

  // Phase 2: softmax. lane (h2,l2): row l2, segment h2 of 256 keys.
  {
    int row = l2, s0 = h2 * 256;
    float mx = -__builtin_inff();
    for (int s = 0; s < 256; ++s) mx = fmaxf(mx, sc[row * TT + s0 + s]);
    mx = fmaxf(mx, __shfl_xor(mx, 16));
    float sum = 0.0f;
    for (int s = 0; s < 256; ++s) {
      float e = __expf(sc[row * TT + s0 + s] - mx);
      sc[row * TT + s0 + s] = e;
      sum += e;
    }
    sum += __shfl_xor(sum, 16);
    float inv = 1.0f / sum;
    for (int s = 0; s < 256; ++s) sc[row * TT + s0 + s] *= inv;
  }
  __syncthreads();

  // Phase 3: out(16 x 64) = probs(16 x 512) @ v(512 x 64)
  v8f oacc[4] = {};
  for (int kt = 0; kt < TT / 32; ++kt) {
    v16bf a;
    #pragma unroll
    for (int e = 0; e < 8; ++e) a[e]     = (__bf16)sc[l2 * TT + kt * 32 + h2 * 16 + e];
    #pragma unroll
    for (int e = 0; e < 8; ++e) a[8 + e] = (__bf16)sc[l2 * TT + kt * 32 + h2 * 16 + 8 + e];
    #pragma unroll
    for (int j = 0; j < 4; ++j) {
      const float* vbase = v + (size_t)(n * TT + kt * 32 + h2 * 16) * DD + head * HD + j * 16 + l2;
      v16bf b;
      #pragma unroll
      for (int e = 0; e < 16; ++e) b[e] = (__bf16)vbase[(size_t)e * DD];
      oacc[j] = wmma_bf16(a, b, oacc[j]);
    }
  }
  #pragma unroll
  for (int j = 0; j < 4; ++j)
    #pragma unroll
    for (int vv = 0; vv < 8; ++vv) {
      int rowl = vv + 8 * h2;
      out[(size_t)(n * TT + mt * 16 + rowl) * DD + head * HD + j * 16 + l2] = oacc[j][vv];
    }
}

// ---------------------------------------------------------------------------
// x[row] = LayerNorm(x[row] + a[arow]) * g + b.   bcast!=0: arow = row / T.
// One 256-thread block per row (D = 512 -> 2 elems/thread).
// ---------------------------------------------------------------------------
__global__ __launch_bounds__(256) void add_ln_kernel(float* __restrict__ x,
                                                     const float* __restrict__ a,
                                                     const float* __restrict__ g,
                                                     const float* __restrict__ b,
                                                     int bcast) {
  int row = blockIdx.x;
  const float* ar = a + (size_t)(bcast ? (row / TT) : row) * DD;
  float* xr = x + (size_t)row * DD;
  int t = threadIdx.x;
  float v0 = xr[t] + ar[t];
  float v1 = xr[t + 256] + ar[t + 256];
  float s = v0 + v1, ss = v0 * v0 + v1 * v1;
  #pragma unroll
  for (int off = 16; off > 0; off >>= 1) {
    s  += __shfl_down(s, off);
    ss += __shfl_down(ss, off);
  }
  __shared__ float sh_s[8], sh_ss[8];
  int wid = t >> 5;
  if ((t & 31) == 0) { sh_s[wid] = s; sh_ss[wid] = ss; }
  __syncthreads();
  if (t == 0) {
    float ts = 0, tss = 0;
    for (int i = 0; i < 8; ++i) { ts += sh_s[i]; tss += sh_ss[i]; }
    sh_s[0] = ts; sh_ss[0] = tss;
  }
  __syncthreads();
  float mu  = sh_s[0] * (1.0f / DD);
  float var = sh_ss[0] * (1.0f / DD) - mu * mu;
  float rinv = rsqrtf(var + 1e-5f);
  xr[t]       = (v0 - mu) * rinv * g[t]       + b[t];
  xr[t + 256] = (v1 - mu) * rinv * g[t + 256] + b[t + 256];
}

// ---------------------------------------------------------------------------
// Host orchestration
// ---------------------------------------------------------------------------
static inline int divup(int a, int b) { return (a + b - 1) / b; }

extern "C" void kernel_launch(void* const* d_in, const int* in_sizes, int n_in,
                              void* d_out, int out_size, void* d_ws, size_t ws_size,
                              hipStream_t stream) {
  const float* features  = (const float*)d_in[0];
  const int*   captions  = (const int*)  d_in[1];
  const float* emb_table = (const float*)d_in[2];
  const float* pos_table = (const float*)d_in[3];
  const float* feat_W    = (const float*)d_in[4];
  const float* feat_b    = (const float*)d_in[5];
  const float* sa_qkv_W  = (const float*)d_in[6];
  const float* sa_qkv_b  = (const float*)d_in[7];
  const float* sa_o_W    = (const float*)d_in[8];
  const float* sa_o_b    = (const float*)d_in[9];
  const float* sa_ln     = (const float*)d_in[10];
  const float* ca_qkv_W  = (const float*)d_in[11];
  const float* ca_qkv_b  = (const float*)d_in[12];
  const float* ca_o_W    = (const float*)d_in[13];
  const float* ca_o_b    = (const float*)d_in[14];
  const float* ca_ln     = (const float*)d_in[15];
  const float* ff_W1     = (const float*)d_in[16];
  const float* ff_b1     = (const float*)d_in[17];
  const float* ff_W2     = (const float*)d_in[18];
  const float* ff_b2     = (const float*)d_in[19];
  const float* ff_ln     = (const float*)d_in[20];
  const float* out_W     = (const float*)d_in[21];
  const float* out_b     = (const float*)d_in[22];

  const int M = NB * TT;           // 4096 token rows
  char* ws = (char*)d_ws;
  size_t off = 0;
  auto alloc = [&](size_t bytes) { void* p = ws + off; off = (off + bytes + 255) & ~(size_t)255; return p; };
  float* x    = (float*)alloc((size_t)M * DD * 4);
  float* qb   = (float*)alloc((size_t)M * DD * 4);
  float* kb   = (float*)alloc((size_t)M * DD * 4);
  float* vb   = (float*)alloc((size_t)M * DD * 4);
  float* att  = (float*)alloc((size_t)M * DD * 4);
  float* hbuf = (float*)alloc((size_t)M * FF * 4);
  __bf16* Ap  = (__bf16*)alloc((size_t)M * FF * 2);   // big enough for K=2048 pack
  __bf16* Bp  = (__bf16*)alloc((size_t)DD * VV * 2);  // big enough for out_W pack
  float* cond = (float*)alloc((size_t)NB * DD * 4);
  float* vc   = (float*)alloc((size_t)NB * DD * 4);
  float* zc   = (float*)alloc((size_t)NB * DD * 4);
  (void)ws_size; (void)in_sizes; (void)n_in; (void)out_size;

  auto packA = [&](const float* A, int Mt, int Kt, int K) {
    int tot = Mt * Kt * 32;
    packA_kernel<<<divup(tot, 256), 256, 0, stream>>>(A, Ap, Mt, Kt, K);
  };
  auto packB = [&](const float* B, int Nt, int Kt, int Nc) {
    int tot = Nt * Kt * 32;
    packB_kernel<<<divup(tot, 256), 256, 0, stream>>>(B, Bp, Nt, Kt, Nc);
  };
  auto gemm = [&](const float* bias, float* C, int Mb, int Nblk, int Kt, int Nc, int relu) {
    int waves = Mb * Nblk;
    gemm_kernel<<<divup(waves * 32, 256), 256, 0, stream>>>(Ap, Bp, bias, C, Mb, Nblk, Kt, Nc, relu);
  };

  const int Mt  = M / 16;          // 256 (pack granularity)
  const int Mb  = M / 32;          // 128 (gemm wave granularity: 32 rows/wave)
  const int KtD = DD / 32;         // 16
  const int KtF = FF / 32;         // 64

  // Embedding + positions
  embed_kernel<<<divup(M * DD, 256), 256, 0, stream>>>(captions, emb_table, pos_table, x);
  // cond = features @ feat_W + feat_b   [8, 512]
  smallmm_kernel<<<divup(NB * DD, 256), 256, 0, stream>>>(features, feat_W, feat_b, cond, NB, DIN, DD);

  for (int l = 0; l < LL; ++l) {
    // ---- causal self-attention ----
    packA(x, Mt, KtD, DD);
    float* qkv[3] = {qb, kb, vb};
    for (int p = 0; p < 3; ++p) {
      packB(sa_qkv_W + (size_t)(l * 3 + p) * DD * DD, DD / 16, KtD, DD);
      gemm(sa_qkv_b + (size_t)(l * 3 + p) * DD, qkv[p], Mb, DD / 64, KtD, DD, 0);
    }
    attn_kernel<<<NB * HH * (TT / 16), 32, 0, stream>>>(qb, kb, vb, att);
    packA(att, Mt, KtD, DD);
    packB(sa_o_W + (size_t)l * DD * DD, DD / 16, KtD, DD);
    gemm(sa_o_b + (size_t)l * DD, att, Mb, DD / 64, KtD, DD, 0);
    add_ln_kernel<<<M, 256, 0, stream>>>(x, att, sa_ln + (size_t)(l * 2) * DD,
                                         sa_ln + (size_t)(l * 2 + 1) * DD, 0);

    // ---- cross-attention: S==1 => softmax==1 => y == V(cond), broadcast over T ----
    smallmm_kernel<<<divup(NB * DD, 256), 256, 0, stream>>>(
        cond, ca_qkv_W + (size_t)(l * 3 + 2) * DD * DD, ca_qkv_b + (size_t)(l * 3 + 2) * DD,
        vc, NB, DD, DD);
    smallmm_kernel<<<divup(NB * DD, 256), 256, 0, stream>>>(
        vc, ca_o_W + (size_t)l * DD * DD, ca_o_b + (size_t)l * DD, zc, NB, DD, DD);
    add_ln_kernel<<<M, 256, 0, stream>>>(x, zc, ca_ln + (size_t)(l * 2) * DD,
                                         ca_ln + (size_t)(l * 2 + 1) * DD, 1);

    // ---- FFN ----
    packA(x, Mt, KtD, DD);
    packB(ff_W1 + (size_t)l * DD * FF, FF / 16, KtD, FF);
    gemm(ff_b1 + (size_t)l * FF, hbuf, Mb, FF / 64, KtD, FF, 1);
    packA(hbuf, Mt, KtF, FF);
    packB(ff_W2 + (size_t)l * FF * DD, DD / 16, KtF, DD);
    gemm(ff_b2 + (size_t)l * DD, att, Mb, DD / 64, KtF, DD, 0);
    add_ln_kernel<<<M, 256, 0, stream>>>(x, att, ff_ln + (size_t)(l * 2) * DD,
                                         ff_ln + (size_t)(l * 2 + 1) * DD, 0);
  }

  // ---- logits: [4096,512] @ [512,32000] + out_b ----
  packA(x, Mt, KtD, DD);
  packB(out_W, VV / 16, KtD, VV);
  gemm(out_b, (float*)d_out, Mb, VV / 64, KtD, VV, 0);
}
